// WordEncoder_59803124630071
// MI455X (gfx1250) — matrix operand order repaired and checked by reading, not compiled, not run on hardware
//
#include <hip/hip_runtime.h>
#include <math.h>
#include <stdint.h>

// Problem constants (match reference)
#define B_  4
#define S_  8192
#define D_  256
#define H_  8
#define W_  64
#define L_  6
#define DH_ 32
#define NW_ 128
#define FF_ 1024
#define M_  (B_*S_)   // 32768 token rows

typedef __attribute__((ext_vector_type(16))) _Float16 v16h;
typedef __attribute__((ext_vector_type(8)))  _Float16 v8h;
typedef __attribute__((ext_vector_type(8)))  float    v8f;
typedef __attribute__((ext_vector_type(4)))  int      v4i;

static_assert(sizeof(v16h) == 32, "v16h must be 32 bytes");
static_assert(sizeof(v8f)  == 32, "v8f must be 32 bytes");
static_assert(sizeof(v4i)  == 16, "v4i must be 16 bytes");

// ---------------------------------------------------------------------------
// CDNA5 async global->LDS staging (ASYNCcnt path), with safe fallback.
#if defined(__has_builtin)
#  if __has_builtin(__builtin_amdgcn_global_load_async_to_lds_b128) && \
      __has_builtin(__builtin_amdgcn_s_wait_asynccnt)
#    define USE_ASYNC_LDS 1
#  endif
#endif
#ifndef USE_ASYNC_LDS
#  define USE_ASYNC_LDS 0
#endif

typedef __attribute__((address_space(1))) v4i global_v4i;  // global memory v4i
typedef __attribute__((address_space(3))) v4i lds_v4i;     // LDS v4i

// Copy 16 bytes global -> LDS. Async variant issues global_load_async_to_lds_b128.
__device__ __forceinline__ void copy16_to_lds(const _Float16* g, _Float16* l) {
#if USE_ASYNC_LDS
  // generic->global: flat address == global address on AMDGPU, reinterpret via int
  global_v4i* gp = (global_v4i*)(uintptr_t)g;
  // generic->LDS: pointee-type change, then addrspacecast to AS3
  lds_v4i* lp = (lds_v4i*)(v4i*)l;
  __builtin_amdgcn_global_load_async_to_lds_b128(gp, lp, 0, 0);
#else
  *(v8h*)l = *(const v8h*)g;
#endif
}

__device__ __forceinline__ void stage_wait() {
#if USE_ASYNC_LDS
  __builtin_amdgcn_s_wait_asynccnt(0);  // s_wait_asynccnt 0
#endif
  __syncthreads();
}

// ---------------------------------------------------------------------------
// WMMA fragment loaders (wave32, V_WMMA_F32_16X16X32_F16; contiguous k order k = 16*(L/16)+e, same for A and B)
//
// A (16x32 f16, M x K): lane L holds row M = L%16.
//   lanes 0-15 : elements 0..15 = K k0+0..15
//   lanes 16-31: elements 0..15 = K k0+16..31
__device__ __forceinline__ v16h load_a_frag(const _Float16* A, int lda, int row0, int k0) {
  const int lane = threadIdx.x & 31;
  const _Float16* p = A + (long long)(row0 + (lane & 15)) * lda + k0 + ((lane >> 4) << 4);
  v8h lo = *(const v8h*)p;
  v8h hi = *(const v8h*)(p + 8);
  v16h r;
#pragma unroll
  for (int i = 0; i < 8; ++i) { r[i] = lo[i]; r[i + 8] = hi[i]; }
  return r;
}

// B (32x16 f16, K x N): lane L holds column N = L%16;
//   lanes 0-15 cover K k0+0..15, lanes 16-31 cover K k0+16..31.
// Bt is [N, K] row-major => one contiguous 32-byte read per lane.
__device__ __forceinline__ v16h load_b_frag_t(const _Float16* Bt, int ldb, int n0, int k0) {
  const int lane = threadIdx.x & 31;
  const _Float16* p = Bt + (long long)(n0 + (lane & 15)) * ldb + k0 + ((lane >> 4) << 4);
  return *(const v16h*)p;
}

// Same B layout but source is row-major [K, N] (strided gather). Used for V in attention.
__device__ __forceinline__ v16h load_b_frag_rm(const _Float16* Bm, int ldb, int k0, int n0) {
  const int lane = threadIdx.x & 31;
  const int n  = n0 + (lane & 15);
  const int kb = k0 + ((lane >> 4) << 4);
  v16h r;
#pragma unroll
  for (int e = 0; e < 16; ++e) r[e] = Bm[(long long)(kb + e) * ldb + n];
  return r;
}

// ---------------------------------------------------------------------------
// Tiled WMMA GEMM with double-buffered async LDS staging.
// C[M,N] = A[M,K] (f16, row-major) x Bt[N,K] (f16, pre-transposed)
// 256 threads = 8 waves; block tile 128x64; wave tile 32x32 (2x2 WMMA frags).
// Per K-step: stage 128x32 A-tile (8KB) + 64x32 B-tile (4KB) into LDS via
// GLOBAL_LOAD_ASYNC_TO_LDS_B128 (3 x b128 per thread), overlap with WMMA.
// mode 0: Cout(f16) = acc                      (QKV projections)
// mode 1: Cres(f32) += acc (+ bias[col])       (O-proj / FFN2 residual add)
// mode 2: Cout(f16) = gelu(acc + bias[col])    (FFN1)
__global__ __launch_bounds__(256)
void gemm_wmma(const _Float16* __restrict__ A, const _Float16* __restrict__ Bt,
               const float* __restrict__ bias, float* __restrict__ Cres,
               _Float16* __restrict__ Cout, int M, int N, int K, int mode)
{
  __shared__ __align__(16) _Float16 sA[2][128 * 32];
  __shared__ __align__(16) _Float16 sB[2][64 * 32];

  const int wave  = threadIdx.x >> 5;
  const int lane  = threadIdx.x & 31;
  const int rblk  = blockIdx.y * 128;
  const int cblk  = blockIdx.x * 64;
  const int rbase = (wave & 3) * 32;
  const int cbase = (wave >> 2) * 32;

  // Per-thread staging descriptors: 768 16B-chunks (512 A + 256 B) / 256 threads.
  const _Float16* gbase[3];
  _Float16*       lbase[3];
  int             lstride[3];
#pragma unroll
  for (int i = 0; i < 3; ++i) {
    const int c = threadIdx.x + i * 256;
    if (c < 512) {                       // A tile: row 0..127, 4 chunks of 8 halfs
      const int row  = c >> 2;
      const int koff = (c & 3) * 8;
      gbase[i]   = A + (long long)(rblk + row) * K + koff;
      lbase[i]   = &sA[0][row * 32 + koff];
      lstride[i] = 128 * 32;
    } else {                             // B tile: n 0..63
      const int n    = (c - 512) >> 2;
      const int koff = ((c - 512) & 3) * 8;
      gbase[i]   = Bt + (long long)(cblk + n) * K + koff;
      lbase[i]   = &sB[0][n * 32 + koff];
      lstride[i] = 64 * 32;
    }
  }

  v8f cc[2][2];
#pragma unroll
  for (int i = 0; i < 2; ++i)
#pragma unroll
    for (int j = 0; j < 2; ++j)
#pragma unroll
      for (int r = 0; r < 8; ++r) cc[i][j][r] = 0.0f;

  // Prologue: stage first K-chunk into buffer 0.
#pragma unroll
  for (int i = 0; i < 3; ++i) copy16_to_lds(gbase[i], lbase[i]);

  int buf = 0;
  for (int k0 = 0; k0 < K; k0 += 32) {
    stage_wait();  // async loads into `buf` complete + visible to all waves

    if (k0 + 32 < K) {  // stage next chunk into the other buffer (overlaps WMMA)
#pragma unroll
      for (int i = 0; i < 3; ++i)
        copy16_to_lds(gbase[i] + k0 + 32, lbase[i] + (buf ^ 1) * lstride[i]);
    }
    if (k0 + 64 < K)  // keep the global_prefetch_b8 path warm for chunk k0+64
      __builtin_prefetch(gbase[0] + k0 + 64, 0, 3);

    const _Float16* cA = sA[buf];
    const _Float16* cB = sB[buf];
    v16h a0 = load_a_frag(cA, 32, rbase,      0);
    v16h a1 = load_a_frag(cA, 32, rbase + 16, 0);
    v16h b0 = load_b_frag_t(cB, 32, cbase,      0);
    v16h b1 = load_b_frag_t(cB, 32, cbase + 16, 0);
    cc[0][0] = __builtin_amdgcn_wmma_f32_16x16x32_f16(false, a0, false, b0, (short)0, cc[0][0], false, false);
    cc[0][1] = __builtin_amdgcn_wmma_f32_16x16x32_f16(false, a0, false, b1, (short)0, cc[0][1], false, false);
    cc[1][0] = __builtin_amdgcn_wmma_f32_16x16x32_f16(false, a1, false, b0, (short)0, cc[1][0], false, false);
    cc[1][1] = __builtin_amdgcn_wmma_f32_16x16x32_f16(false, a1, false, b1, (short)0, cc[1][1], false, false);

    __syncthreads();  // all waves done reading `buf` before it is restaged
    buf ^= 1;
  }

#pragma unroll
  for (int i = 0; i < 2; ++i)
#pragma unroll
    for (int j = 0; j < 2; ++j)
#pragma unroll
      for (int r = 0; r < 8; ++r) {
        const int row = rblk + rbase + i * 16 + r + ((lane >> 4) << 3);
        const int col = cblk + cbase + j * 16 + (lane & 15);
        const long long idx = (long long)row * N + col;
        float vv = cc[i][j][r];
        if (mode == 0) {
          Cout[idx] = (_Float16)vv;
        } else if (mode == 1) {
          if (bias) vv += bias[col];
          Cres[idx] += vv;
        } else {
          vv += bias[col];
          const float x3 = vv * vv * vv;
          const float g  = 0.5f * vv * (1.0f + tanhf(0.7978845608028654f * (vv + 0.044715f * x3)));
          Cout[idx] = (_Float16)g;
        }
      }
}

// ---------------------------------------------------------------------------
// Fused local attention for one (window, head, batch). 256 threads = 8 waves.
// LDS: Kctx/Vctx (192x32 f16), scores (64x192 f16), mask addend (192 f32) = ~49 KB.
__global__ __launch_bounds__(256)
void attn_kernel(const _Float16* __restrict__ q, const _Float16* __restrict__ k,
                 const _Float16* __restrict__ v, const int* __restrict__ amask,
                 _Float16* __restrict__ out)
{
  __shared__ __align__(16) _Float16 sK[192 * DH_];
  __shared__ __align__(16) _Float16 sV[192 * DH_];
  __shared__ __align__(16) _Float16 sS[W_ * 192];
  __shared__ float sAdd[192];

  const int win  = blockIdx.x;
  const int h    = blockIdx.y;
  const int b    = blockIdx.z;
  const int lane = threadIdx.x & 31;
  const int wave = threadIdx.x >> 5;

  // Stage K/V context (prev | own | next window), zero + -inf mask for invalid keys.
  for (int idx = threadIdx.x; idx < 192 * DH_; idx += 256) {
    const int j    = idx >> 5;          // context position 0..191
    const int dh   = idx & 31;
    const int wctx = win - 1 + (j >> 6);
    int valid = (wctx >= 0) && (wctx < NW_);
    const int spos = wctx * W_ + (j & 63);
    if (valid) valid = (amask[b * S_ + spos] != 0);
    _Float16 kv = (_Float16)0.f, vv = (_Float16)0.f;
    if (valid) {
      const long long base = ((long long)(b * S_ + spos)) * D_ + h * DH_ + dh;
      kv = k[base]; vv = v[base];
    }
    sK[idx] = kv; sV[idx] = vv;
    if (dh == 0) sAdd[j] = valid ? 0.0f : -30000.0f;
  }
  __syncthreads();

  // scores[64,192] = Q[64,32] * Kctx^T : 48 16x16 tiles, K-dim = DH = 32 (one WMMA each)
  const float scale = 0.17677669529663687f;  // 1/sqrt(32)
  for (int tt = wave; tt < 48; tt += 8) {
    const int tm = tt & 3, tn = tt >> 2;
    v16h a  = load_a_frag(q, D_, b * S_ + win * W_ + tm * 16, h * DH_);
    v16h bf = load_b_frag_t(sK, DH_, tn * 16, 0);
    v8f c = {};
    c = __builtin_amdgcn_wmma_f32_16x16x32_f16(false, a, false, bf, (short)0, c, false, false);
#pragma unroll
    for (int r = 0; r < 8; ++r) {
      const int rr = tm * 16 + r + ((lane >> 4) << 3);
      const int nn = tn * 16 + (lane & 15);
      sS[rr * 192 + nn] = (_Float16)(c[r] * scale + sAdd[nn]);
    }
  }
  __syncthreads();

  // Row softmax over 192 keys (one thread per query row; waves 0-1).
  if (threadIdx.x < W_) {
    _Float16* sr = sS + threadIdx.x * 192;
    float mx = -1e30f;
    for (int j = 0; j < 192; ++j) mx = fmaxf(mx, (float)sr[j]);
    float sum = 0.f;
    for (int j = 0; j < 192; ++j) sum += __expf((float)sr[j] - mx);
    const float inv = 1.0f / sum;
    for (int j = 0; j < 192; ++j) sr[j] = (_Float16)(__expf((float)sr[j] - mx) * inv);
  }
  __syncthreads();

  // out[64,32] = attn[64,192] * Vctx[192,32] : 8 tiles (4x2), 6 WMMAs each.
  {
    const int tm = wave & 3, tn = wave >> 2;
    v8f acc = {};
#pragma unroll
    for (int kc = 0; kc < 192; kc += 32) {
      v16h a  = load_a_frag(sS, 192, tm * 16, kc);
      v16h bf = load_b_frag_rm(sV, DH_, kc, tn * 16);
      acc = __builtin_amdgcn_wmma_f32_16x16x32_f16(false, a, false, bf, (short)0, acc, false, false);
    }
#pragma unroll
    for (int r = 0; r < 8; ++r) {
      const int row = win * W_ + tm * 16 + r + ((lane >> 4) << 3);
      const int col = h * DH_ + tn * 16 + (lane & 15);
      out[((long long)(b * S_ + row)) * D_ + col] = (_Float16)acc[r];
    }
  }
}

// ---------------------------------------------------------------------------
// LayerNorm over D=256: one wave32 per row, 8 elements/lane, shfl_xor reduction.
__global__ __launch_bounds__(256)
void ln_kernel(const float* __restrict__ x, const float* __restrict__ g,
               const float* __restrict__ bta, _Float16* __restrict__ out16,
               float* __restrict__ out32)
{
  const int row  = blockIdx.x * 8 + (threadIdx.x >> 5);
  const int lane = threadIdx.x & 31;
  const float* xr = x + (long long)row * D_;
  float vals[8];
  float s = 0.f;
#pragma unroll
  for (int i = 0; i < 8; ++i) { vals[i] = xr[lane + i * 32]; s += vals[i]; }
#pragma unroll
  for (int off = 16; off > 0; off >>= 1) s += __shfl_xor(s, off, 32);
  const float mean = s * (1.0f / D_);
  float vs = 0.f;
#pragma unroll
  for (int i = 0; i < 8; ++i) { const float d = vals[i] - mean; vs += d * d; }
#pragma unroll
  for (int off = 16; off > 0; off >>= 1) vs += __shfl_xor(vs, off, 32);
  const float rstd = rsqrtf(vs * (1.0f / D_) + 1e-5f);
#pragma unroll
  for (int i = 0; i < 8; ++i) {
    const int c = lane + i * 32;
    const float y = (vals[i] - mean) * rstd * g[c] + bta[c];
    if (out16) out16[(long long)row * D_ + c] = (_Float16)y;
    else       out32[(long long)row * D_ + c] = y;
  }
}

// ---------------------------------------------------------------------------
__global__ void embed_kernel(const int* __restrict__ ids, const float* __restrict__ emb,
                             const float* __restrict__ pos, float* __restrict__ x)
{
  const long long t = (long long)blockIdx.x * blockDim.x + threadIdx.x;  // M*64 float4 slots
  const int dv = (int)(t & 63);
  const int m  = (int)(t >> 6);
  const int s  = m & (S_ - 1);
  const int id = ids[m];
  const float4 e  = ((const float4*)(emb + (long long)id * D_))[dv];
  const float4 pp = ((const float4*)(pos + (long long)s  * D_))[dv];
  ((float4*)(x + (long long)m * D_))[dv] =
      make_float4(e.x + pp.x, e.y + pp.y, e.z + pp.z, e.w + pp.w);
}

// Convert f32 weight [K,N] -> f16 transposed [N,K] (so B-fragments are contiguous).
__global__ void wcvt_kernel(const float* __restrict__ w, _Float16* __restrict__ wt,
                            int K, int N)
{
  const long long t = (long long)blockIdx.x * blockDim.x + threadIdx.x;  // K*N
  const int n  = (int)(t % N);
  const int kk = (int)(t / N);
  wt[(long long)n * K + kk] = (_Float16)w[t];
}

__global__ void zero_kernel(float4* __restrict__ p)
{
  const long long t = (long long)blockIdx.x * blockDim.x + threadIdx.x;
  p[t] = make_float4(0.f, 0.f, 0.f, 0.f);
}

// Per-batch prefix scan of word_start -> target row (-1 if not a word start).
__global__ __launch_bounds__(256)
void scan_kernel(const int* __restrict__ ws, int* __restrict__ tgt)
{
  __shared__ int ssum[256];
  __shared__ int sbase[256];
  const int b = blockIdx.x, t = threadIdx.x;
  const int* w = ws + b * S_;
  const int s0 = t * (S_ / 256);
  int c = 0;
  for (int i = 0; i < S_ / 256; ++i) c += (w[s0 + i] > 0);
  ssum[t] = c;
  __syncthreads();
  if (t == 0) { int run = 0; for (int i = 0; i < 256; ++i) { sbase[i] = run; run += ssum[i]; } }
  __syncthreads();
  int run = sbase[t];
  int* tg = tgt + b * S_;
  for (int i = 0; i < S_ / 256; ++i) {
    if (w[s0 + i] > 0) { tg[s0 + i] = run; ++run; }
    else               { tg[s0 + i] = -1; }
  }
}

__global__ void scatter_kernel(const float* __restrict__ y, const int* __restrict__ tgt,
                               float* __restrict__ out)
{
  const long long t = (long long)blockIdx.x * blockDim.x + threadIdx.x;  // M*64 float4 slots
  const int dv = (int)(t & 63);
  const int m  = (int)(t >> 6);
  const int tg = tgt[m];
  if (tg >= 0) {
    const int b = m / S_;
    ((float4*)(out + ((long long)(b * S_ + tg)) * D_))[dv] =
        ((const float4*)(y + (long long)m * D_))[dv];
  }
}

// ---------------------------------------------------------------------------
extern "C" void kernel_launch(void* const* d_in, const int* in_sizes, int n_in,
                              void* d_out, int out_size, void* d_ws, size_t ws_size,
                              hipStream_t stream)
{
  (void)in_sizes; (void)n_in; (void)out_size; (void)ws_size;

  const int*   ids    = (const int*)d_in[0];
  const int*   amask  = (const int*)d_in[1];
  const int*   wstart = (const int*)d_in[2];
  const float* emb    = (const float*)d_in[3];
  const float* pos    = (const float*)d_in[4];
  const float* ln1g   = (const float*)d_in[5];
  const float* ln1b   = (const float*)d_in[6];
  const float* wq     = (const float*)d_in[7];
  const float* wk     = (const float*)d_in[8];
  const float* wvp    = (const float*)d_in[9];
  const float* wo     = (const float*)d_in[10];
  const float* ln2g   = (const float*)d_in[11];
  const float* ln2b   = (const float*)d_in[12];
  const float* w1     = (const float*)d_in[13];
  const float* b1     = (const float*)d_in[14];
  const float* w2     = (const float*)d_in[15];
  const float* b2     = (const float*)d_in[16];
  const float* lnfg   = (const float*)d_in[17];
  const float* lnfb   = (const float*)d_in[18];

  // Workspace carve (~218 MB total)
  char* p = (char*)d_ws;
  float*    x   = (float*)p;     p += (size_t)M_ * D_  * 4;   // residual stream f32
  _Float16* xh  = (_Float16*)p;  p += (size_t)M_ * D_  * 2;   // LN'd activations f16
  _Float16* qb  = (_Float16*)p;  p += (size_t)M_ * D_  * 2;
  _Float16* kb  = (_Float16*)p;  p += (size_t)M_ * D_  * 2;
  _Float16* vb  = (_Float16*)p;  p += (size_t)M_ * D_  * 2;
  _Float16* ao  = (_Float16*)p;  p += (size_t)M_ * D_  * 2;   // attention output f16
  _Float16* hf  = (_Float16*)p;  p += (size_t)M_ * FF_ * 2;   // FFN hidden f16
  float*    y   = (float*)p;     p += (size_t)M_ * D_  * 4;   // final LN output f32
  _Float16* wts = (_Float16*)p;  p += (size_t)L_ * (4 * D_ * D_ + 2 * D_ * FF_) * 2;
  int*      tgt = (int*)p;       p += (size_t)M_ * 4;

  const size_t WPL = 4 * (size_t)D_ * D_ + 2 * (size_t)D_ * FF_;  // halfs per layer
  const dim3 blk(256);

  // 1) weight convert + transpose to f16 [N,K]
  for (int l = 0; l < L_; ++l) {
    _Float16* wl  = wts + (size_t)l * WPL;
    _Float16* wqT = wl;
    _Float16* wkT = wl + 1 * D_ * D_;
    _Float16* wvT = wl + 2 * D_ * D_;
    _Float16* woT = wl + 3 * D_ * D_;
    _Float16* w1T = wl + 4 * D_ * D_;
    _Float16* w2T = wl + 4 * D_ * D_ + D_ * FF_;
    wcvt_kernel<<<(D_ * D_) / 256, blk, 0, stream>>>(wq  + (size_t)l * D_ * D_,  wqT, D_,  D_);
    wcvt_kernel<<<(D_ * D_) / 256, blk, 0, stream>>>(wk  + (size_t)l * D_ * D_,  wkT, D_,  D_);
    wcvt_kernel<<<(D_ * D_) / 256, blk, 0, stream>>>(wvp + (size_t)l * D_ * D_,  wvT, D_,  D_);
    wcvt_kernel<<<(D_ * D_) / 256, blk, 0, stream>>>(wo  + (size_t)l * D_ * D_,  woT, D_,  D_);
    wcvt_kernel<<<(D_ * FF_) / 256, blk, 0, stream>>>(w1 + (size_t)l * D_ * FF_, w1T, D_,  FF_);
    wcvt_kernel<<<(D_ * FF_) / 256, blk, 0, stream>>>(w2 + (size_t)l * FF_ * D_, w2T, FF_, D_);
  }

  // 2) embeddings
  embed_kernel<<<(M_ * (D_ / 4)) / 256, blk, 0, stream>>>(ids, emb, pos, x);

  // 3) transformer layers
  for (int l = 0; l < L_; ++l) {
    _Float16* wl  = wts + (size_t)l * WPL;
    _Float16* wqT = wl;
    _Float16* wkT = wl + 1 * D_ * D_;
    _Float16* wvT = wl + 2 * D_ * D_;
    _Float16* woT = wl + 3 * D_ * D_;
    _Float16* w1T = wl + 4 * D_ * D_;
    _Float16* w2T = wl + 4 * D_ * D_ + D_ * FF_;

    ln_kernel<<<M_ / 8, blk, 0, stream>>>(x, ln1g + l * D_, ln1b + l * D_, xh, nullptr);

    gemm_wmma<<<dim3(D_ / 64, M_ / 128), blk, 0, stream>>>(xh, wqT, nullptr, nullptr, qb, M_, D_, D_, 0);
    gemm_wmma<<<dim3(D_ / 64, M_ / 128), blk, 0, stream>>>(xh, wkT, nullptr, nullptr, kb, M_, D_, D_, 0);
    gemm_wmma<<<dim3(D_ / 64, M_ / 128), blk, 0, stream>>>(xh, wvT, nullptr, nullptr, vb, M_, D_, D_, 0);

    attn_kernel<<<dim3(NW_, H_, B_), blk, 0, stream>>>(qb, kb, vb, amask, ao);

    gemm_wmma<<<dim3(D_ / 64, M_ / 128), blk, 0, stream>>>(ao, woT, nullptr, x, nullptr, M_, D_, D_, 1);

    ln_kernel<<<M_ / 8, blk, 0, stream>>>(x, ln2g + l * D_, ln2b + l * D_, xh, nullptr);

    gemm_wmma<<<dim3(FF_ / 64, M_ / 128), blk, 0, stream>>>(xh, w1T, b1 + (size_t)l * FF_, nullptr, hf, M_, FF_, D_, 2);
    gemm_wmma<<<dim3(D_ / 64, M_ / 128), blk, 0, stream>>>(hf, w2T, b2 + (size_t)l * D_, x, nullptr, M_, D_, FF_, 1);
  }

  // 4) final LN (f32 out)
  ln_kernel<<<M_ / 8, blk, 0, stream>>>(x, lnfg, lnfb, nullptr, y);

  // 5) word-start gather: zero output, scan, scatter
  zero_kernel<<<(M_ * (D_ / 4)) / 256, blk, 0, stream>>>((float4*)d_out);
  scan_kernel<<<B_, blk, 0, stream>>>(wstart, tgt);
  scatter_kernel<<<(M_ * (D_ / 4)) / 256, blk, 0, stream>>>(y, tgt, (float*)d_out);
}
